// BlockDiagonalGRU_80882824119117
// MI455X (gfx1250) — compile-verified, matrix operand order checked by reading, not run on hardware
//
#include <hip/hip_runtime.h>
#include <hip/hip_bf16.h>

// ---------------------------------------------------------------------------
// Block-diagonal GRU for MI455X (gfx1250, wave32, WMMA).
//   bf16 v_wmma_f32_16x16x32_bf16 GEMMs, f32 accumulate.
//   A-operands (x/h tiles, M=32) staged in dynamic LDS (padded, conflict-free),
//   each wave computes a 32x32 C-region (2 b-tiles x 2 f-tiles):
//     per 4 WMMAs -> 4 global b128 (B) + 4 ds b128 (A): balanced L0/LDS paths.
//   Gate buffers: summed for r/z regions, split only for the n-gate region.
// ---------------------------------------------------------------------------

typedef __bf16 bf16_t;
typedef __attribute__((ext_vector_type(16))) __bf16 v16bf;
typedef __attribute__((ext_vector_type(8)))  float  v8f;

#define T_STEPS 64
#define BATCH   256
#define IN      1024
#define HID     4096
#define NB      8
#define BLK     512      // HID / NB
#define G3      12288    // 3 * HID
#define GPB     1536     // 3 * BLK (per-block gate rows)

#define MT      32       // batch rows per workgroup
#define XPAD    1032     // IN  + 8 bf16 pad -> 2064B row stride (4-bank rotate)
#define HPAD    520      // BLK + 8 bf16 pad -> 1040B row stride
#define SMEM_BYTES ((MT * XPAD + MT * HPAD) * 2)   // 99,328 B (<320KB WGP LDS)

// ---- one-time f32 -> bf16 conversion ---------------------------------------
__global__ void cvt_f32_bf16(const float* __restrict__ src,
                             bf16_t* __restrict__ dst, int n) {
    int i = blockIdx.x * blockDim.x + threadIdx.x;
    if (i < n) dst[i] = (bf16_t)src[i];
}

// ---- h state init: copy h_0 to f32 working state + bf16 mirror -------------
__global__ void init_h(const float* __restrict__ h0,
                       float* __restrict__ h, bf16_t* __restrict__ hb, int n) {
    int i = blockIdx.x * blockDim.x + threadIdx.x;
    if (i < n) { float v = h0[i]; h[i] = v; hb[i] = (bf16_t)v; }
}

// ---- bf16 WMMA fragment load (A or B, K-major) -----------------------------
// ISA 16-bit operand layout, lane (mn = l&15, khalf = l>>4):
//   frag[j]   <- K = khalf*8 + j        (j=0..7)  -> 16B load @ k0 + khalf*8
//   frag[j+8] <- K = 16 + khalf*8 + j   (j=0..7)  -> 16B load @ k0 + 16 + khalf*8
__device__ __forceinline__ v16bf load_frag(const bf16_t* row_k0, int khalf) {
    union { uint4 u[2]; v16bf v; } t;
    t.u[0] = *reinterpret_cast<const uint4*>(row_k0 + khalf * 8);
    t.u[1] = *reinterpret_cast<const uint4*>(row_k0 + 16 + khalf * 8);
    return t.v;
}

// ---- per-step fused gate GEMM ----------------------------------------------
// For f <  2H: bufG[b,f]  = xg + hg + W_i_b[f] + b_h[f]          (r/z gates)
// For f >= 2H: bufG[b,f]  = xg + W_i_b[f]                        (x n-gate)
//              bufHn[b,j] = hg + b_h[f],  j = f - 2H             (h n-gate)
// xg: K=1024 vs W_i;  hg: K=512 vs W_h block (nblk = f/1536).
// Grid: x = G3/256 = 48 (8 waves x 2 f-tiles), y = BATCH/32 = 8.
__global__ void __launch_bounds__(256)
gemm_step(const bf16_t* __restrict__ xt,    // [BATCH][IN]  (timestep slice)
          const bf16_t* __restrict__ hb,    // [BATCH][HID]
          const bf16_t* __restrict__ Wi,    // [G3][IN]
          const bf16_t* __restrict__ Wh,    // [G3][BLK] (flattened (NB,1536,512))
          const float*  __restrict__ Wib,   // [G3]
          const float*  __restrict__ bh,    // [G3]
          float* __restrict__ bufG,         // [BATCH][G3]
          float* __restrict__ bufHn)        // [BATCH][HID]
{
    extern __shared__ __align__(16) char smem[];
    bf16_t* lds_x = (bf16_t*)smem;                       // [MT][XPAD]
    bf16_t* lds_h = (bf16_t*)(smem + MT * XPAD * 2);     // [MT][HPAD]

    const int tid   = threadIdx.x;
    const int lane  = tid & 31;
    const int wave  = tid >> 5;
    const int f0    = blockIdx.x * 256;               // 256-col WG tile
    const int bb0   = blockIdx.y * MT;                // batch tile (32 rows)
    const int mn    = lane & 15;
    const int khalf = lane >> 4;
    const int nblk  = f0 / GPB;                       // uniform: 1536 = 6*256

    // ---- stage A tiles (x rows, h block rows) into LDS ----
    #pragma unroll
    for (int c = tid; c < MT * (IN / 8); c += 256) {        // 32 x 1024 bf16
        int row = c >> 7, col = (c & 127) << 3;
        *(uint4*)&lds_x[row * XPAD + col] =
            *(const uint4*)&xt[(size_t)(bb0 + row) * IN + col];
    }
    #pragma unroll
    for (int c = tid; c < MT * (BLK / 8); c += 256) {       // 32 x 512 bf16
        int row = c >> 6, col = (c & 63) << 3;
        *(uint4*)&lds_h[row * HPAD + col] =
            *(const uint4*)&hb[(size_t)(bb0 + row) * HID + nblk * BLK + col];
    }
    __syncthreads();

    const int fA = f0 + wave * 32;                    // two adjacent f-tiles
    const int fB = fA + 16;

    v8f accX[2][2] = {};   // [b-tile][f-tile]
    v8f accH[2][2] = {};

    // ---- input projection, K = 1024 (A from LDS, B from L2-resident Wi) ----
    const bf16_t* browA = Wi + (size_t)(fA + mn) * IN;
    const bf16_t* browB = Wi + (size_t)(fB + mn) * IN;
    const bf16_t* lxr0  = &lds_x[mn * XPAD];
    const bf16_t* lxr1  = &lds_x[(mn + 16) * XPAD];
    #pragma unroll 2
    for (int k = 0; k < IN; k += 32) {
        v16bf a0 = load_frag(lxr0 + k, khalf);        // ds_load_b128 x2
        v16bf a1 = load_frag(lxr1 + k, khalf);
        v16bf b0 = load_frag(browA + k, khalf);       // global_load_b128 x2
        v16bf b1 = load_frag(browB + k, khalf);
        accX[0][0] = __builtin_amdgcn_wmma_f32_16x16x32_bf16(
                         false, a0, false, b0, (short)0, accX[0][0], false, false);
        accX[0][1] = __builtin_amdgcn_wmma_f32_16x16x32_bf16(
                         false, a0, false, b1, (short)0, accX[0][1], false, false);
        accX[1][0] = __builtin_amdgcn_wmma_f32_16x16x32_bf16(
                         false, a1, false, b0, (short)0, accX[1][0], false, false);
        accX[1][1] = __builtin_amdgcn_wmma_f32_16x16x32_bf16(
                         false, a1, false, b1, (short)0, accX[1][1], false, false);
    }

    // ---- block-diagonal recurrent part, K = 512 ----
    const bf16_t* crowA = Wh + (size_t)(fA + mn) * BLK;
    const bf16_t* crowB = Wh + (size_t)(fB + mn) * BLK;
    const bf16_t* lhr0  = &lds_h[mn * HPAD];
    const bf16_t* lhr1  = &lds_h[(mn + 16) * HPAD];
    #pragma unroll 2
    for (int s = 0; s < BLK; s += 32) {
        v16bf a0 = load_frag(lhr0 + s, khalf);
        v16bf a1 = load_frag(lhr1 + s, khalf);
        v16bf b0 = load_frag(crowA + s, khalf);
        v16bf b1 = load_frag(crowB + s, khalf);
        accH[0][0] = __builtin_amdgcn_wmma_f32_16x16x32_bf16(
                         false, a0, false, b0, (short)0, accH[0][0], false, false);
        accH[0][1] = __builtin_amdgcn_wmma_f32_16x16x32_bf16(
                         false, a0, false, b1, (short)0, accH[0][1], false, false);
        accH[1][0] = __builtin_amdgcn_wmma_f32_16x16x32_bf16(
                         false, a1, false, b0, (short)0, accH[1][0], false, false);
        accH[1][1] = __builtin_amdgcn_wmma_f32_16x16x32_bf16(
                         false, a1, false, b1, (short)0, accH[1][1], false, false);
    }

    // ---- epilogue: C/D layout -> VGPR r holds (M = r + 8*khalf, N = lane&15)
    #pragma unroll
    for (int ft = 0; ft < 2; ++ft) {
        int   f   = (ft ? fB : fA) + mn;
        float wib = Wib[f];
        float bhv = bh[f];
        #pragma unroll
        for (int bt = 0; bt < 2; ++bt) {
            int mbase = bb0 + bt * 16 + khalf * 8;
            if (f < 2 * HID) {             // r/z gates: only the sum is needed
                #pragma unroll
                for (int r = 0; r < 8; ++r) {
                    bufG[(size_t)(mbase + r) * G3 + f] =
                        accX[bt][ft][r] + accH[bt][ft][r] + wib + bhv;
                }
            } else {                       // n-gate: keep xg and hg separate
                #pragma unroll
                for (int r = 0; r < 8; ++r) {
                    bufG [(size_t)(mbase + r) * G3  + f] = accX[bt][ft][r] + wib;
                    bufHn[(size_t)(mbase + r) * HID + (f - 2 * HID)] =
                        accH[bt][ft][r] + bhv;
                }
            }
        }
    }
}

// ---- per-step pointwise gate update ----------------------------------------
__global__ void __launch_bounds__(256)
pointwise_step(const float* __restrict__ bufG, const float* __restrict__ bufHn,
               float* __restrict__ h, bf16_t* __restrict__ hb,
               float* __restrict__ out)    // out = d_out + t*BATCH*HID
{
    int i = blockIdx.x * blockDim.x + threadIdx.x;   // [0, BATCH*HID)
    int b = i >> 12;            // / HID
    int j = i & (HID - 1);
    size_t base = (size_t)b * G3;

    float r = bufG[base + j];
    r = 1.0f / (1.0f + __expf(-r));
    float z = bufG[base + HID + j];
    z = 1.0f / (1.0f + __expf(-z));
    float xn = bufG[base + 2 * HID + j];
    float hn = bufHn[(size_t)b * HID + j];
    float n  = tanhf(xn + r * hn);

    float ho = h[i];
    float hv = (1.0f - z) * n + z * ho;
    h[i]  = hv;
    hb[i] = (bf16_t)hv;
    out[i] = hv;
}

// ---------------------------------------------------------------------------
extern "C" void kernel_launch(void* const* d_in, const int* in_sizes, int n_in,
                              void* d_out, int out_size, void* d_ws, size_t ws_size,
                              hipStream_t stream) {
    const float* x   = (const float*)d_in[0];   // [T][B][IN]
    const float* h0  = (const float*)d_in[1];   // [B][HID]
    const float* Wi  = (const float*)d_in[2];   // [G3][IN]
    const float* Wib = (const float*)d_in[3];   // [G3]
    const float* Wh  = (const float*)d_in[4];   // [NB][1536][512]
    const float* bh  = (const float*)d_in[5];   // [G3]
    float* out = (float*)d_out;                 // [T][B][HID] ++ [B][HID]

    // workspace carve-up (~91 MB total), 256B-aligned slices
    char* ws = (char*)d_ws;
    size_t off = 0;
    auto alloc = [&](size_t bytes) -> void* {
        void* p = (void*)(ws + off);
        off += (bytes + 255) & ~(size_t)255;
        return p;
    };
    bf16_t* x_b   = (bf16_t*)alloc((size_t)T_STEPS * BATCH * IN * 2);  // 32 MB
    bf16_t* Wi_b  = (bf16_t*)alloc((size_t)G3 * IN * 2);               // 24 MB
    bf16_t* Wh_b  = (bf16_t*)alloc((size_t)G3 * BLK * 2);              // 12 MB
    float*  h_f   = (float*) alloc((size_t)BATCH * HID * 4);           //  4 MB
    bf16_t* h_b   = (bf16_t*)alloc((size_t)BATCH * HID * 2);           //  2 MB
    float*  bufG  = (float*) alloc((size_t)BATCH * G3 * 4);            // 12 MB
    float*  bufHn = (float*) alloc((size_t)BATCH * HID * 4);           //  4 MB

    // one-time conversions (weights become L2-resident bf16)
    {
        int n;
        n = T_STEPS * BATCH * IN;
        cvt_f32_bf16<<<(n + 255) / 256, 256, 0, stream>>>(x, x_b, n);
        n = G3 * IN;
        cvt_f32_bf16<<<(n + 255) / 256, 256, 0, stream>>>(Wi, Wi_b, n);
        n = G3 * BLK;
        cvt_f32_bf16<<<(n + 255) / 256, 256, 0, stream>>>(Wh, Wh_b, n);
        n = BATCH * HID;
        init_h<<<(n + 255) / 256, 256, 0, stream>>>(h0, h_f, h_b, n);
    }

    dim3 ggrid(G3 / 256, BATCH / MT);   // 48 x 8 workgroups, 8 waves each
    int pw_blocks = (BATCH * HID) / 256;

    for (int t = 0; t < T_STEPS; ++t) {
        gemm_step<<<ggrid, 256, SMEM_BYTES, stream>>>(
            x_b + (size_t)t * BATCH * IN, h_b, Wi_b, Wh_b, Wib, bh, bufG, bufHn);
        pointwise_step<<<pw_blocks, 256, 0, stream>>>(
            bufG, bufHn, h_f, h_b, out + (size_t)t * BATCH * HID);
    }

    // h_T tail of the output
    hipMemcpyAsync(out + (size_t)T_STEPS * BATCH * HID, h_f,
                   (size_t)BATCH * HID * sizeof(float),
                   hipMemcpyDeviceToDevice, stream);
}
